// ContextualRPEIdxQ_45775761440795
// MI455X (gfx1250) — matrix operand order repaired and checked by reading, not compiled, not run on hardware
//
#include <hip/hip_runtime.h>

typedef __attribute__((ext_vector_type(16))) _Float16 v16h;
typedef __attribute__((ext_vector_type(8)))  float    v8f;

#define NHEAD 8
#define HPD   16
#define LQ    32
#define NB    4    // queries (n) per block; 8 waves = 4 n x 2 l-tiles

__global__ __launch_bounds__(256) void crpe_wmma_kernel(
    const float* __restrict__ relpos,   // [N, 32, 3]
    const float* __restrict__ qfeat,    // [N, 8, 48]
    const float* __restrict__ scaling,  // [1]
    const float* __restrict__ tabx,     // [353, 8, 16]
    const float* __restrict__ taby,     // [401, 8, 16]
    const float* __restrict__ tabz,     // [21, 8, 16]
    float* __restrict__ out,            // [N, 32, 8]
    int N)
{
    __shared__ int   s_idx[NB][LQ][3];          // gathered table row ids
    __shared__ float s_qf[NB][NHEAD][48];       // query features for the n-block
    __shared__ float s_out[8][16][NHEAD];       // per-wave D staging for coalesced stores

    const int tid   = threadIdx.x;
    const int nbase = blockIdx.x * NB;

    // ---------------- Phase 1a: quantize relpos -> indices, prefetch hot table rows
    for (int t = tid; t < NB * LQ * 3; t += 256) {
        int nl   = t / 3, a = t - nl * 3;
        int nloc = nl / LQ, l = nl - nloc * LQ;
        int n    = nbase + nloc; if (n >= N) n = N - 1;   // clamp (N%NB==0 normally)
        float md = (a == 0) ? 70.4f : ((a == 1) ? 80.0f : 4.0f);
        int   ts = (a == 0) ? 353   : ((a == 1) ? 401   : 21);
        const float* tp = (a == 0) ? tabx : ((a == 1) ? taby : tabz);
        float v  = relpos[((size_t)n * LQ + l) * 3 + a];
        int  id  = (int)floorf((v + md) * 2.5f);          // 1/QUAN_SIZE = 2.5
        id = min(max(id, 0), ts - 1);
        s_idx[nloc][l][a] = id;
        __builtin_prefetch(tp + (size_t)id * (NHEAD * HPD), 0, 0); // warm WGP$/L2
    }

    // ---------------- Phase 1b: stage qf block into LDS (float4, coalesced)
    for (int t = tid; t < NB * (NHEAD * 48 / 4); t += 256) {
        int nloc = t / 96, w = t - nloc * 96;             // 96 float4 per n
        int n    = nbase + nloc; if (n >= N) n = N - 1;
        ((float4*)&s_qf[0][0][0])[t] =
            ((const float4*)(qfeat + (size_t)n * NHEAD * 48))[w];
    }
    __syncthreads();

    // ---------------- Phase 2: per-wave WMMA over K=(h,d), block-diagonal B
    const int wave  = tid >> 5;
    const int lane  = tid & 31;
    const int nloc  = wave >> 1;
    const int ltile = wave & 1;
    const int nreal = nbase + nloc;
    const float s   = scaling[0];

    const int  m    = lane & 15;          // A: row M (l within tile); B: column h'
    const bool lo   = lane < 16;
    const int  off4 = lo ? 0 : 4;         // h = 2c vs h = 2c+1 (float4 units)
    const int  l    = ltile * 16 + m;

    // Per axis, this lane supplies B column data (with hk == m) in at most one
    // chunk: low-half lanes with even m at chunk m/2, high-half lanes with odd
    // m at chunk m>>1.  Load that qf slice once per axis; per chunk just select.
    int myc = -1;
    if ((m < NHEAD) && (lo == ((m & 1) == 0))) myc = m >> 1;

    const v16h Bzero = {};
    v8f acc = {};
    #pragma unroll
    for (int a = 0; a < 3; ++a) {
        const float* tab = (a == 0) ? tabx : ((a == 1) ? taby : tabz);
        const int r = s_idx[nloc][l][a];
        const float4* rp = (const float4*)(tab + (size_t)r * (NHEAD * HPD));

        // hoisted B source operand for this axis (zero for non-supplying lanes)
        v16h Bq = {};
        if (myc >= 0) {
            const float* qh = &s_qf[nloc][m][a * HPD];
            #pragma unroll
            for (int j = 0; j < 16; ++j) Bq[j] = (_Float16)qh[j];
        }

        #pragma unroll
        for (int c = 0; c < 4; ++c) {     // K chunk: global k = 32c + klocal, h = 2c + klocal/16
            // A k order chosen here (contiguous, same as B): lanes 0-15 hold K 0-15, lanes 16-31 hold K 16-31
            float4 f0 = rp[8 * c + off4];          // d = 0..3
            float4 f1 = rp[8 * c + off4 + 1];      // d = 4..7
            float4 f2 = rp[8 * c + off4 + 2];      // d = 8..11
            float4 f3 = rp[8 * c + off4 + 3];      // d = 12..15
            v16h A;
            A[0]=(_Float16)f0.x;  A[1]=(_Float16)f0.y;  A[2]=(_Float16)f0.z;  A[3]=(_Float16)f0.w;
            A[4]=(_Float16)f1.x;  A[5]=(_Float16)f1.y;  A[6]=(_Float16)f1.z;  A[7]=(_Float16)f1.w;
            A[8]=(_Float16)f2.x;  A[9]=(_Float16)f2.y;  A[10]=(_Float16)f2.z; A[11]=(_Float16)f2.w;
            A[12]=(_Float16)f3.x; A[13]=(_Float16)f3.y; A[14]=(_Float16)f3.z; A[15]=(_Float16)f3.w;

            // B k order chosen here (contiguous, same as A): lanes 0-15 hold K 0-15, lanes 16-31 hold K 16-31;
            // lane = column h'.  Block-diagonal: nonzero only where h(k) == h'.
            v16h B = (c == myc) ? Bq : Bzero;

            acc = __builtin_amdgcn_wmma_f32_16x16x32_f16(
                      false, A, false, B, (short)0, acc, false, false);
        }
    }

    // ---------------- Stage D (f32 C/D layout) into LDS, then coalesced float4 stores
    {
        const int mb = lo ? 0 : 8;        // lane<16: rows 0-7, lane>=16: rows 8-15
        if (m < NHEAD) {
            #pragma unroll
            for (int r8 = 0; r8 < 8; ++r8) s_out[wave][mb + r8][m] = acc[r8] * s;
        }
    }
    __syncthreads();
    if (nreal < N) {
        float4 v = ((const float4*)&s_out[wave][0][0])[lane];
        float4* op = (float4*)(out + ((size_t)nreal * LQ + ltile * 16) * NHEAD);
        op[lane] = v;
    }
}

extern "C" void kernel_launch(void* const* d_in, const int* in_sizes, int n_in,
                              void* d_out, int out_size, void* d_ws, size_t ws_size,
                              hipStream_t stream) {
    const float* relpos = (const float*)d_in[0];
    const float* qfeat  = (const float*)d_in[1];
    const float* scal   = (const float*)d_in[2];
    // d_in[3] = query_batch_cnt (unused by the reference math)
    const float* tabx   = (const float*)d_in[4];
    const float* taby   = (const float*)d_in[5];
    const float* tabz   = (const float*)d_in[6];
    float* out = (float*)d_out;

    const int N = in_sizes[0] / (LQ * 3);
    dim3 grid((N + NB - 1) / NB), block(256);
    hipLaunchKernelGGL(crpe_wmma_kernel, grid, block, 0, stream,
                       relpos, qfeat, scal, tabx, taby, tabz, out, N);
}
